// GridEncoder_23081154249396
// MI455X (gfx1250) — matrix operand order repaired and checked by reading, not compiled, not run on hardware
//
#include <hip/hip_runtime.h>
#include <hip/hip_bf16.h>

// ---------------------------------------------------------------------------
// Problem constants (from reference)
// ---------------------------------------------------------------------------
#define HW_IN     96
#define CH        128          // 16 pieces * 8 filters
#define H0        94           // after grouped 3x3 VALID
#define H1        92           // after conv #1
#define H2        90           // after conv #2
#define NPIX0     (H0*H0)      // 8836
#define NPIX1     (H1*H1)      // 8464
#define NPIX2     (H2*H2)      // 8100
#define FEAT      (NPIX2*CH)   // 1,036,800
#define OUT_DIM   256
#define NCHUNK    64           // GEMV column chunks per row

#define WPACK_ELEMS (36 * 8 * 32 * 16)       // 147,456 bf16
#define WPACK_BYTES (WPACK_ELEMS * 2)        // 294,912 B  (< 320 KB WGP LDS)
#define CONV_WAVES  8

typedef __attribute__((ext_vector_type(16))) __bf16 v16bf;
typedef __attribute__((ext_vector_type(8)))  float  v8f;
typedef __attribute__((ext_vector_type(4)))  float  v4f;

__device__ __forceinline__ __bf16 f2bf(float f) {
    unsigned u = __builtin_bit_cast(unsigned, f);
    unsigned r = (u + 0x7FFFu + ((u >> 16) & 1u)) >> 16;     // RNE
    return __builtin_bit_cast(__bf16, (unsigned short)r);
}

// Hardware A-slot K permutation for 16-bit WMMA A (16x32):
//   contiguous per-lane element e maps to hw-K: lanes0-15 {0..7,16..23},
//   lanes16-31 {8..15,24..31}. We load A contiguously, so B packing places
//   the channel A holds at hw-K k:
__device__ __host__ __forceinline__ int chan_of_k(int k) {
    if (k < 8)  return k;
    if (k < 16) return k + 8;
    if (k < 24) return k - 8;
    return k;
}

// ---------------------------------------------------------------------------
// 1) Pack conv_w (Cout=128, Cin=128, 3, 3) f32 -> bf16 wave32 B-tiles.
// Layout: [kstep(36)][co_tile(8)][lane(32)][e(16)]
//   kstep = (di*3+dj)*4 + cb   (cb = 32-channel block)
//   B layout (contiguous k, same order as the A load): lane<16 -> N=lane, K=e ; lane>=16 -> N=lane-16, K=16+e
// ---------------------------------------------------------------------------
__global__ void pack_w_kernel(const float* __restrict__ conv_w,
                              __bf16* __restrict__ wpack) {
    int idx = blockIdx.x * blockDim.x + threadIdx.x;
    if (idx >= WPACK_ELEMS) return;
    int e     = idx & 15;
    int lane  = (idx >> 4) & 31;
    int t     = (idx >> 9) & 7;     // co tile
    int kstep = idx >> 12;          // 0..35
    int cb = kstep & 3;
    int t9 = kstep >> 2;
    int di = t9 / 3, dj = t9 % 3;
    int co = t * 16 + (lane & 15);
    int khw = (lane < 16) ? e : (16 + e);
    int ci = cb * 32 + khw;
    float v = conv_w[((co * CH) + ci) * 9 + di * 3 + dj];
    wpack[idx] = f2bf(v);
}

// ---------------------------------------------------------------------------
// 2) One-hot + grouped conv + bias + relu -> y0 (HWC bf16, 94x94x128)
// ---------------------------------------------------------------------------
__global__ __launch_bounds__(128)
void gconv_kernel(const int* __restrict__ x,
                  const float* __restrict__ cat_w,   // (16,8,1,3,3) = c*9+t
                  const float* __restrict__ cat_b,   // (16,8) = c
                  __bf16* __restrict__ y0) {
    int pix = blockIdx.x;                  // 0..NPIX0-1
    int oi = pix / H0, oj = pix % H0;
    __shared__ int xs[9];
    if (threadIdx.x < 9)
        xs[threadIdx.x] = x[(oi + threadIdx.x / 3) * HW_IN + (oj + threadIdx.x % 3)];
    __syncthreads();
    int c = threadIdx.x;                   // 0..127
    int p = c >> 3;                        // piece
    float s = cat_b[c];
    #pragma unroll
    for (int t = 0; t < 9; ++t)
        if (xs[t] == p) s += cat_w[c * 9 + t];
    s = s > 0.f ? s : 0.f;
    y0[pix * CH + c] = f2bf(s);
}

// ---------------------------------------------------------------------------
// 3) Dense 3x3 VALID conv via implicit GEMM, v_wmma_f32_16x16x32_bf16.
// 8 waves per block; the whole 294 KB packed-B matrix is staged into LDS
// with the gfx1250 async global->LDS path (ASYNCcnt), then every wave
// computes one 16-pixel x 128-channel tile feeding WMMA B from ds_load.
// ---------------------------------------------------------------------------
template <int HIN, int HOUT>
__global__ __launch_bounds__(CONV_WAVES * 32)
void conv_wmma_kernel(const __bf16* __restrict__ in,     // HWC, HIN x HIN x 128
                      const __bf16* __restrict__ wpack,  // global packed B
                      const float* __restrict__ bias,
                      __bf16* __restrict__ out_bf,       // HWC or null
                      float* __restrict__ out_f32) {     // CHW or null
    constexpr int Win  = HIN;
    constexpr int Wout = HOUT;
    constexpr int npix = HOUT * HOUT;
    constexpr int ntiles = (npix + 15) / 16;

    extern __shared__ char smem[];
    __bf16* lds_w = (__bf16*)smem;                       // WPACK_BYTES

    // ---- async stage: global wpack -> LDS (18,432 x 16B chunks) ----
    {
        const int tid = threadIdx.x;
        unsigned lbase = (unsigned)(unsigned long long)(&lds_w[0]); // low 32 = LDS offset
        const char* g = (const char*)wpack;
        for (int i = tid; i < WPACK_BYTES / 16; i += CONV_WAVES * 32) {
            unsigned l = lbase + (unsigned)i * 16u;
            unsigned long long ga = (unsigned long long)(g + (size_t)i * 16);
            asm volatile("global_load_async_to_lds_b128 %0, %1, off"
                         :: "v"(l), "v"(ga) : "memory");
        }
        asm volatile("s_wait_asynccnt 0x0" ::: "memory");
        __syncthreads();
    }

    const int lane = threadIdx.x & 31;
    const int wave = threadIdx.x >> 5;
    int tile = blockIdx.x * CONV_WAVES + wave;
    if (tile >= ntiles) tile = ntiles - 1;   // duplicate last tile: same-value stores, keeps EXEC uniform

    const int mrow = lane & 15;
    const int ksel = lane >> 4;              // which 16-ch half of the K=32 slab
    int pix = tile * 16 + mrow;
    int p = pix < npix ? pix : npix - 1;     // clamp loads; stores are predicated
    int oi = p / Wout, oj = p - oi * Wout;

    v8f acc[8] = {};
    const v16bf* wv = (const v16bf*)lds_w;

    #pragma unroll
    for (int t9 = 0; t9 < 9; ++t9) {
        const __bf16* arow =
            in + ((size_t)(oi + t9 / 3) * Win + (oj + t9 % 3)) * CH + ksel * 16;
        #pragma unroll
        for (int cb = 0; cb < 4; ++cb) {
            v16bf a = *(const v16bf*)(arow + cb * 32);
            const v16bf* bb = wv + (size_t)((t9 * 4 + cb) * 8) * 32 + lane;
            #pragma unroll
            for (int t = 0; t < 8; ++t) {
                acc[t] = __builtin_amdgcn_wmma_f32_16x16x32_bf16(
                    false, a, false, bb[t * 32], (short)0, acc[t], false, false);
            }
        }
    }

    // C/D layout: vgpr r, lane -> M = r + (lane>>4)*8, N = lane&15
    const int nlo = lane & 15;
    #pragma unroll
    for (int t = 0; t < 8; ++t) {
        int ch = t * 16 + nlo;
        float bs = bias[ch];
        #pragma unroll
        for (int r = 0; r < 8; ++r) {
            int pm = tile * 16 + r + ksel * 8;
            float v = acc[t][r] + bs;
            v = v > 0.f ? v : 0.f;
            if (pm < npix) {
                if (out_bf)  out_bf[(size_t)pm * CH + ch] = f2bf(v);
                if (out_f32) out_f32[(size_t)ch * npix + pm] = v;  // CHW flatten
            }
        }
    }
}

// ---------------------------------------------------------------------------
// 4) GEMV: out[row] = lin_w[row,:] . flat  — pure HBM-bandwidth bound
// (1.06 GB of fp32 weights -> ~45us floor at 23.3 TB/s). Non-temporal loads
// keep lin_w out of L2 so the 4.1 MB flat vector stays resident.
// ---------------------------------------------------------------------------
__global__ __launch_bounds__(256)
void gemv_kernel(const float* __restrict__ W,
                 const float* __restrict__ v,
                 float* __restrict__ partials) {
    const int FEAT4 = FEAT / 4;            // 259200
    const int C4 = FEAT4 / NCHUNK;         // 4050
    int row = blockIdx.y, chunk = blockIdx.x, tid = threadIdx.x;
    const v4f* W4 = (const v4f*)W + (size_t)row * FEAT4 + (size_t)chunk * C4;
    const v4f* V4 = (const v4f*)v + (size_t)chunk * C4;
    float acc = 0.f;
    for (int i = tid; i < C4; i += 256) {
        v4f w = __builtin_nontemporal_load(W4 + i);
        v4f a = V4[i];
        acc += w.x * a.x + w.y * a.y + w.z * a.z + w.w * a.w;
    }
    __shared__ float red[256];
    red[tid] = acc;
    __syncthreads();
    #pragma unroll
    for (int s = 128; s > 0; s >>= 1) {
        if (tid < s) red[tid] += red[tid + s];
        __syncthreads();
    }
    if (tid == 0) partials[(size_t)row * NCHUNK + chunk] = red[0];
}

__global__ __launch_bounds__(256)
void reduce_kernel(const float* __restrict__ partials,
                   const float* __restrict__ lin_b,
                   float* __restrict__ out) {
    int o = threadIdx.x;
    float s = lin_b[o];
    for (int c = 0; c < NCHUNK; ++c) s += partials[(size_t)o * NCHUNK + c];
    out[o] = s;
}

// ---------------------------------------------------------------------------
// Host launcher
// ---------------------------------------------------------------------------
extern "C" void kernel_launch(void* const* d_in, const int* in_sizes, int n_in,
                              void* d_out, int out_size, void* d_ws, size_t ws_size,
                              hipStream_t stream) {
    const int*   x      = (const int*)  d_in[0];
    const float* cat_w  = (const float*)d_in[1];
    const float* cat_b  = (const float*)d_in[2];
    const float* conv_w = (const float*)d_in[3];
    const float* conv_b = (const float*)d_in[4];
    const float* lin_w  = (const float*)d_in[5];
    const float* lin_b  = (const float*)d_in[6];
    float* out = (float*)d_out;

    char* ws = (char*)d_ws;
    size_t off = 0;
    __bf16* y0 = (__bf16*)(ws + off);  off += (size_t)NPIX0 * CH * 2;   // 2,262,016
    __bf16* y1 = (__bf16*)(ws + off);  off += (size_t)NPIX1 * CH * 2;   // 2,166,784
    float*  flat = (float*)(ws + off); off += (size_t)FEAT * 4;         // 4,147,200
    __bf16* wpack = (__bf16*)(ws + off); off += (size_t)WPACK_BYTES;
    float*  partials = (float*)(ws + off); off += (size_t)OUT_DIM * NCHUNK * 4;

    // 1) pack conv weights into WMMA B layout (bf16)
    pack_w_kernel<<<(WPACK_ELEMS + 255) / 256, 256, 0, stream>>>(conv_w, wpack);

    // 2) one-hot grouped conv -> y0 (94x94x128 bf16 HWC)
    gconv_kernel<<<NPIX0, 128, 0, stream>>>(x, cat_w, cat_b, y0);

    // 3) conv #1: 94 -> 92, bf16 out.  8 waves/block, B staged in LDS.
    {
        constexpr int ntiles = (NPIX1 + 15) / 16;               // 529
        int blocks = (ntiles + CONV_WAVES - 1) / CONV_WAVES;    // 67
        conv_wmma_kernel<H0, H1>
            <<<blocks, CONV_WAVES * 32, WPACK_BYTES, stream>>>(
                y0, wpack, conv_b, y1, nullptr);
    }

    // 4) conv #2: 92 -> 90, f32 CHW out (reference flatten order)
    {
        constexpr int ntiles = (NPIX2 + 15) / 16;               // 507
        int blocks = (ntiles + CONV_WAVES - 1) / CONV_WAVES;    // 64
        conv_wmma_kernel<H1, H2>
            <<<blocks, CONV_WAVES * 32, WPACK_BYTES, stream>>>(
                y1, wpack, conv_b, nullptr, flat);
    }

    // 5) GEMV over 1.06 GB of lin_w (HBM-bandwidth bound)
    dim3 g(NCHUNK, OUT_DIM);
    gemv_kernel<<<g, 256, 0, stream>>>(lin_w, flat, partials);

    // 6) deterministic final reduction + bias
    reduce_kernel<<<1, OUT_DIM, 0, stream>>>(partials, lin_b, out);
}
